// RnnModelMITBIH_85375359910327
// MI455X (gfx1250) — matrix operand order repaired and checked by reading, not compiled, not run on hardware
//
#include <hip/hip_runtime.h>

typedef _Float16 h16;
typedef __attribute__((ext_vector_type(8)))  _Float16 v8h;
typedef __attribute__((ext_vector_type(16))) _Float16 v16h;
typedef __attribute__((ext_vector_type(8)))  float    v8f;

#define TSEQ 9
#define SROW 72                      // halves per activation row; 144B stride (16B aligned)
#define HROW 40                      // halves per h-state row; 80B stride (16B aligned)
#define BUFSZ (TSEQ * 16 * SROW)     // one activation buffer, in halves
#define HOFF (2 * BUFSZ)             // h-state area start, in halves
#define SHSZ (2 * BUFSZ + 2 * 16 * HROW)

// ---------------- WMMA helpers ----------------

__device__ __forceinline__ v8f wmma16(v16h a, v16h b, v8f c) {
  // D = A(16x32 f16) * B(32x16 f16) + C(16x16 f32)
  return __builtin_amdgcn_wmma_f32_16x16x32_f16(false, a, false, b, (short)0, c,
                                                false, false);
}

// A fragment (16x32, f16) from an LDS tile [16 rows][stride halves].
// CDNA5 A layout: lane (m = lane&15, hi = lane>>4) holds elements
//   a[0..7]  = K = kbase + 8*hi + j
//   a[8..15] = K = kbase + 16 + 8*hi + (j-8)
// -> two contiguous 16B (ds_load_b128) loads per lane.
__device__ __forceinline__ v16h load_A(const h16* sh, int tile_off, int stride, int m,
                                       int kbase, int hi) {
  const h16* r = sh + tile_off + m * stride + kbase + 8 * hi;
  v8h lo = *reinterpret_cast<const v8h*>(r);
  v8h hh = *reinterpret_cast<const v8h*>(r + 16);
  return __builtin_shufflevector(lo, hh, 0, 1, 2, 3, 4, 5, 6, 7,
                                 8, 9, 10, 11, 12, 13, 14, 15);
}

// B fragment (32x16, f16) built from a row-major f32 weight matrix W[nrows][ldw]:
// B[k][n] = W[n][k].  Lane (n = lane&15, hi = lane>>4) holds b[j] = K = kbase+8*hi+j (j<8), kbase+16+8*hi+(j-8) (j>=8).
// Zero outside klim / nrows (for padded K and padded N).
__device__ __forceinline__ v16h load_B(const float* W, int ldw, int ncol, int nrows,
                                       int kbase, int klim, int hi) {
  v16h b;
#pragma unroll
  for (int j = 0; j < 16; ++j) {
    int k = kbase + (j < 8 ? 8 * hi + j : 16 + 8 * hi + (j - 8));
    float v = (ncol < nrows && k < klim) ? W[ncol * ldw + k] : 0.0f;
    b[j] = (h16)v;
  }
  return b;
}

// Scatter a C/D fragment (lane holds col n, rows m = v+8*hi) into an LDS tile as f16.
template <bool RELU>
__device__ __forceinline__ void store_D(const v8f& c, h16* sh, int tile_off, int stride,
                                        int chbase, int n, int hi) {
#pragma unroll
  for (int v = 0; v < 8; ++v) {
    float x = c[v];
    if (RELU) x = fmaxf(x, 0.0f);
    sh[tile_off + (v + 8 * hi) * stride + chbase + n] = (h16)x;
  }
}

// ---------------- one bidirectional ReLU-RNN layer ----------------
// Both directions advance concurrently in one pass over t (fwd at tt, bwd at 8-tt):
// 2x the independent WMMA chains per iteration fills WMMA->WMMA RAW hazard slots
// and halves loop/sync overhead.
// Activations live in the shared arena `sh`; in_off/out_off are *integer* offsets
// (phi on ints, not pointers, so all accesses stay addrspace(3) -> ds_* ops).
// out channels: dir0 -> [0,H), dir1 -> [H,2H)  (matches reference concat order)
// h-states at sh[HOFF + d*16*HROW], zero-padded so the recurrent K=32 read is valid
// for H=16, and so h_{-1}=0 needs no special case.
template <int H, int IN>
__device__ __forceinline__ void bidir_layer(h16* sh, int in_off, int out_off,
                                            const float* wih, const float* whh,
                                            const float* bih, const float* bhh,
                                            int lane) {
  constexpr int NH = H / 16;          // N tiles
  constexpr int KC = (IN + 31) / 32;  // K chunks of the input projection
  const int n = lane & 15;
  const int hi = lane >> 4;

  // zero both direction h-states
  for (int i = lane; i < (2 * 16 * HROW) / 2; i += 32)
    reinterpret_cast<float*>(sh + HOFF)[i] = 0.0f;
  __syncthreads();

  float bias[2][NH];
  v16h Bih[2][NH][KC];
  v16h Bhh[2][NH];
#pragma unroll
  for (int d = 0; d < 2; ++d) {
    const float* Wih = wih + d * H * IN;
    const float* Whh = whh + d * H * H;
#pragma unroll
    for (int a = 0; a < NH; ++a) {
      bias[d][a] = bih[d * H + a * 16 + n] + bhh[d * H + a * 16 + n];
#pragma unroll
      for (int kc = 0; kc < KC; ++kc)
        Bih[d][a][kc] = load_B(Wih, IN, a * 16 + n, H, kc * 32, IN, hi);
      Bhh[d][a] = load_B(Whh, H, a * 16 + n, H, 0, H, hi);
    }
  }

#pragma unroll 1
  for (int tt = 0; tt < TSEQ; ++tt) {
    const int t0 = tt;             // forward position
    const int t1 = TSEQ - 1 - tt;  // backward position
    const int in0 = in_off + t0 * 16 * SROW;
    const int in1 = in_off + t1 * 16 * SROW;

    v8f c[2][NH];
#pragma unroll
    for (int d = 0; d < 2; ++d)
#pragma unroll
      for (int a = 0; a < NH; ++a)
#pragma unroll
        for (int v = 0; v < 8; ++v) c[d][a][v] = bias[d][a];

    // input projections: x_t @ Wih^T (both directions)
#pragma unroll
    for (int kc = 0; kc < KC; ++kc) {
      v16h Ax0 = load_A(sh, in0, SROW, n, kc * 32, hi);
      v16h Ax1 = load_A(sh, in1, SROW, n, kc * 32, hi);
#pragma unroll
      for (int a = 0; a < NH; ++a) {
        c[0][a] = wmma16(Ax0, Bih[0][a][kc], c[0][a]);
        c[1][a] = wmma16(Ax1, Bih[1][a][kc], c[1][a]);
      }
    }
    // recurrences: h_{t-1} @ Whh^T   (h-state zero at tt==0)
    v16h Ah0 = load_A(sh, HOFF, HROW, n, 0, hi);
    v16h Ah1 = load_A(sh, HOFF + 16 * HROW, HROW, n, 0, hi);
#pragma unroll
    for (int a = 0; a < NH; ++a) {
      c[0][a] = wmma16(Ah0, Bhh[0][a], c[0][a]);
      c[1][a] = wmma16(Ah1, Bhh[1][a], c[1][a]);
    }

    // relu; write layer output (disjoint channel ranges) and recurrent states
    const int out0 = out_off + t0 * 16 * SROW;
    const int out1 = out_off + t1 * 16 * SROW;
#pragma unroll
    for (int a = 0; a < NH; ++a) {
      store_D<true>(c[0][a], sh, out0, SROW, a * 16, n, hi);
      store_D<true>(c[0][a], sh, HOFF, HROW, a * 16, n, hi);
      store_D<true>(c[1][a], sh, out1, SROW, H + a * 16, n, hi);
      store_D<true>(c[1][a], sh, HOFF + 16 * HROW, HROW, a * 16, n, hi);
    }
    __syncthreads();
  }
  __syncthreads();
}

// ---------------- full model kernel: one wave per 16-row batch tile ----------------

__global__ void __launch_bounds__(32) rnn_mitbih_kernel(
    const float* __restrict__ x,
    const float* b0_wih0, const float* b0_wih, const float* b0_whh,
    const float* b0_bih, const float* b0_bhh,
    const float* b1_wih0, const float* b1_wih, const float* b1_whh,
    const float* b1_bih, const float* b1_bhh,
    const float* b2_wih0, const float* b2_wih, const float* b2_whh,
    const float* b2_bih, const float* b2_bhh,
    const float* b3_wih0, const float* b3_wih, const float* b3_whh,
    const float* b3_bih, const float* b3_bhh,
    const float* hw1, const float* hb1, const float* hw2, const float* hb2,
    float* __restrict__ out) {
  __shared__ __align__(16) h16 sh[SHSZ];

  const int lane = threadIdx.x;
  const int n = lane & 15;
  const int hi = lane >> 4;
  const long batch0 = (long)blockIdx.x * 16;

  // zero buffer 0: gives the K-padding 21->32 for layer0 and the 187->189 seq padding
  for (int i = lane; i < BUFSZ / 2; i += 32)
    reinterpret_cast<float*>(sh)[i] = 0.0f;
  __syncthreads();

  // stage x: 16 rows x 187 f32 -> f16 chunks [t][m][k<21]   (coalesced over col)
  for (int i = lane; i < 16 * 187; i += 32) {
    int row = i / 187;
    int col = i - row * 187;
    int t = col / 21;
    int k = col - t * 21;
    float v = x[(batch0 + row) * 187 + col];
    sh[(t * 16 + row) * SROW + k] = (h16)v;
  }
  __syncthreads();

  int bin = 0;
  int bout = BUFSZ;

  // ---- block0: H=32; layer inputs 21, 64, 64
  bidir_layer<32, 21>(sh, bin, bout, b0_wih0, b0_whh, b0_bih, b0_bhh, lane);
  { int t_ = bin; bin = bout; bout = t_; }
#pragma unroll 1
  for (int l = 1; l < 3; ++l) {
    bidir_layer<32, 64>(sh, bin, bout, b0_wih + (l - 1) * 2 * 32 * 64,
                        b0_whh + l * 2 * 32 * 32, b0_bih + l * 2 * 32,
                        b0_bhh + l * 2 * 32, lane);
    { int t_ = bin; bin = bout; bout = t_; }
  }

  // ---- block1: H=16; layer inputs 64, 32, 32, 32
  bidir_layer<16, 64>(sh, bin, bout, b1_wih0, b1_whh, b1_bih, b1_bhh, lane);
  { int t_ = bin; bin = bout; bout = t_; }
#pragma unroll 1
  for (int l = 1; l < 4; ++l) {
    bidir_layer<16, 32>(sh, bin, bout, b1_wih + (l - 1) * 2 * 16 * 32,
                        b1_whh + l * 2 * 16 * 16, b1_bih + l * 2 * 16,
                        b1_bhh + l * 2 * 16, lane);
    { int t_ = bin; bin = bout; bout = t_; }
  }

  // ---- blocks 2 & 3: H=16; all layer inputs 32
#pragma unroll 1
  for (int blk = 0; blk < 2; ++blk) {
    const float* wih0 = blk ? b3_wih0 : b2_wih0;
    const float* wih = blk ? b3_wih : b2_wih;
    const float* whh = blk ? b3_whh : b2_whh;
    const float* bi = blk ? b3_bih : b2_bih;
    const float* bh = blk ? b3_bhh : b2_bhh;
#pragma unroll 1
    for (int l = 0; l < 4; ++l) {
      const float* w = (l == 0) ? wih0 : (wih + (l - 1) * 2 * 16 * 32);
      bidir_layer<16, 32>(sh, bin, bout, w, whh + l * 2 * 16 * 16,
                          bi + l * 2 * 16, bh + l * 2 * 16, lane);
      { int t_ = bin; bin = bout; bout = t_; }
    }
  }

  // ---- head: feat[16][288] = sh[bin + ...][t][m][ch<32];  288->64 relu -> 5
  const int zoff = bout;  // free buffer
  v8f z[4];
#pragma unroll
  for (int a = 0; a < 4; ++a) {
    float bb = hb1[a * 16 + n];
#pragma unroll
    for (int v = 0; v < 8; ++v) z[a][v] = bb;
  }
#pragma unroll 1
  for (int t = 0; t < TSEQ; ++t) {  // K chunk t covers feature cols t*32..t*32+31
    v16h Af = load_A(sh, bin + t * 16 * SROW, SROW, n, 0, hi);
#pragma unroll
    for (int a = 0; a < 4; ++a) {
      v16h Bw = load_B(hw1, 288, a * 16 + n, 64, t * 32, 288, hi);
      z[a] = wmma16(Af, Bw, z[a]);
    }
  }
#pragma unroll
  for (int a = 0; a < 4; ++a) store_D<true>(z[a], sh, zoff, SROW, a * 16, n, hi);
  __syncthreads();

  v8f o;
  {
    float bb = (n < 5) ? hb2[n] : 0.0f;
#pragma unroll
    for (int v = 0; v < 8; ++v) o[v] = bb;
  }
#pragma unroll
  for (int kc = 0; kc < 2; ++kc) {
    v16h Az = load_A(sh, zoff, SROW, n, kc * 32, hi);
    v16h Bw = load_B(hw2, 64, n, 5, kc * 32, 64, hi);
    o = wmma16(Az, Bw, o);
  }
  if (n < 5) {
#pragma unroll
    for (int v = 0; v < 8; ++v) out[(batch0 + v + 8 * hi) * 5 + n] = o[v];
  }
}

extern "C" void kernel_launch(void* const* d_in, const int* in_sizes, int n_in,
                              void* d_out, int out_size, void* d_ws, size_t ws_size,
                              hipStream_t stream) {
  (void)n_in; (void)d_ws; (void)ws_size; (void)out_size;
  const float* x = (const float*)d_in[0];
  const float* b0_wih0 = (const float*)d_in[1];
  const float* b0_wih = (const float*)d_in[2];
  const float* b0_whh = (const float*)d_in[3];
  const float* b0_bih = (const float*)d_in[4];
  const float* b0_bhh = (const float*)d_in[5];
  const float* b1_wih0 = (const float*)d_in[6];
  const float* b1_wih = (const float*)d_in[7];
  const float* b1_whh = (const float*)d_in[8];
  const float* b1_bih = (const float*)d_in[9];
  const float* b1_bhh = (const float*)d_in[10];
  const float* b2_wih0 = (const float*)d_in[11];
  const float* b2_wih = (const float*)d_in[12];
  const float* b2_whh = (const float*)d_in[13];
  const float* b2_bih = (const float*)d_in[14];
  const float* b2_bhh = (const float*)d_in[15];
  const float* b3_wih0 = (const float*)d_in[16];
  const float* b3_wih = (const float*)d_in[17];
  const float* b3_whh = (const float*)d_in[18];
  const float* b3_bih = (const float*)d_in[19];
  const float* b3_bhh = (const float*)d_in[20];
  const float* hw1 = (const float*)d_in[21];
  const float* hb1 = (const float*)d_in[22];
  const float* hw2 = (const float*)d_in[23];
  const float* hb2 = (const float*)d_in[24];
  float* out = (float*)d_out;

  const int Bsz = in_sizes[0] / 187;  // 131072
  const int nblocks = Bsz / 16;       // one wave per 16-row batch tile
  rnn_mitbih_kernel<<<dim3(nblocks), dim3(32), 0, stream>>>(
      x, b0_wih0, b0_wih, b0_whh, b0_bih, b0_bhh,
      b1_wih0, b1_wih, b1_whh, b1_bih, b1_bhh,
      b2_wih0, b2_wih, b2_whh, b2_bih, b2_bhh,
      b3_wih0, b3_wih, b3_whh, b3_bih, b3_bhh,
      hw1, hb1, hw2, hb2, out);
}
